// RankMixerBlock_50766513439288
// MI455X (gfx1250) — matrix operand order repaired and checked by reading, not compiled, not run on hardware
//
#include <hip/hip_runtime.h>
#include <hip/hip_bf16.h>

// ---------------- problem constants ----------------
#define BB   4096          // batch
#define TT   8             // tokens
#define DD   512           // dim
#define HH   1024          // hidden
#define ROWS (BB * TT)     // 32768 LN rows

// fused-FFN tiling
#define BM      64                 // rows per block (halves per-row L2 weight traffic vs 32)
#define NTHR    512                // 16 waves: wm in 0..3, wn in 0..3
#define SA_STR  (DD + 16)          // 528 bf16 (1056B row; pad == TDM pad_amount of 8 DWORDs)
#define SH_STR  (HH + 16)          // 1040
#define SA_BYTES ((size_t)BM * SA_STR * 2)                 // 67584
#define SH_BYTES ((size_t)BM * SH_STR * 2)                 // 133120
#define RED_OFF  (SA_BYTES + SH_BYTES)                     // 200704
#define SMEM_BYTES (RED_OFF + 2 * BM * sizeof(float))      // 201216

// ws layout (bytes)
#define WS_XLN_OFF 0
#define WS_W1S_OFF ((size_t)ROWS * DD * 2)                 // 33,554,432
#define WS_W2S_OFF (WS_W1S_OFF + (size_t)TT * DD * HH * 2) // +8,388,608

typedef __attribute__((ext_vector_type(16))) __bf16 v16bf;
typedef __attribute__((ext_vector_type(8)))  float  v8f;
typedef unsigned int u32x4 __attribute__((ext_vector_type(4)));
typedef int          i32x8 __attribute__((ext_vector_type(8)));
typedef int          i32x4 __attribute__((ext_vector_type(4)));

union FragU { v16bf v; uint4 q[2]; };

__device__ __forceinline__ __bf16 to_bf16(float f) {
    union { float f; unsigned u; } a; a.f = f;
    unsigned r = (a.u + 0x7FFFu + ((a.u >> 16) & 1u)) >> 16;
    union { unsigned short s; __bf16 h; } b; b.s = (unsigned short)r;
    return b.h;
}
__device__ __forceinline__ float from_bf16(__bf16 h) {
    union { unsigned short s; __bf16 h; } b; b.h = h;
    union { unsigned u; float f; } a; a.u = ((unsigned)b.s) << 16;
    return a.f;
}
__device__ __forceinline__ float gelu_exact(float v) {
    return 0.5f * v * (1.0f + erff(v * 0.70710678118654752f));
}

// A fragment (16-bit A 16x32 layout): lane (hi,l16) reads row m, two contiguous
// 8-element k-chunks at k0+hi*8 and k0+16+hi*8  -> two ds_load_b128.
__device__ __forceinline__ v16bf load_a_frag(const __bf16* s, int stride, int m, int k0, int hi) {
    FragU f;
    const char* base = (const char*)(s + (size_t)m * stride + k0 + hi * 8);
    f.q[0] = *(const uint4*)(base);
    f.q[1] = *(const uint4*)(base + 32);
    return f.v;
}
// B fragment: pre-swizzled in global memory, 32 contiguous bytes per lane (L2-resident).
__device__ __forceinline__ v16bf load_b_frag(const __bf16* w, size_t ofs_elems) {
    FragU f;
    const uint4* p = (const uint4*)(w + ofs_elems);
    f.q[0] = p[0];
    f.q[1] = p[1];
    return f.v;
}

// ------------------------------------------------------------------
// Kernel A: fp32 -> bf16 weight conversion into WMMA B-fragment order.
// B layout (16-bit, 16x16x32): matches A table order,
// component c -> k = kt*32 + hi*8 + (c&7) + (c>>3)*16, n = nt*16 + (lane&15).
// ------------------------------------------------------------------
__global__ void rm_swizzle_w(const float* __restrict__ W1, const float* __restrict__ W2,
                             __bf16* __restrict__ w1s, __bf16* __restrict__ w2s) {
    int idx = blockIdx.x * 256 + threadIdx.x;
    const int NW = TT * DD * HH;                 // 4,194,304
    if (idx < NW) {
        int c    = idx & 15;
        int lane = (idx >> 4) & 31;
        int nt   = (idx >> 9) & 63;              // N=1024 -> 64 n-tiles
        int kt   = (idx >> 15) & 15;             // K=512  -> 16 k-tiles
        int t    = idx >> 19;
        int hi = lane >> 4, l16 = lane & 15;
        int k = kt * 32 + hi * 8 + (c & 7) + ((c >> 3) << 4);
        int n = nt * 16 + l16;
        w1s[idx] = to_bf16(W1[((size_t)t * DD + k) * HH + n]);
    } else {
        idx -= NW;
        int c    = idx & 15;
        int lane = (idx >> 4) & 31;
        int nt   = (idx >> 9) & 31;              // N=512  -> 32 n-tiles
        int kt   = (idx >> 14) & 31;             // K=1024 -> 32 k-tiles
        int t    = idx >> 19;
        int hi = lane >> 4, l16 = lane & 15;
        int k = kt * 32 + hi * 8 + (c & 7) + ((c >> 3) << 4);
        int n = nt * 16 + l16;
        w2s[idx] = to_bf16(W2[((size_t)t * HH + k) * DD + n]);
    }
}

// ------------------------------------------------------------------
// Kernel B: token mixing + residual + LayerNorm, emit bf16 x_ln.
// ------------------------------------------------------------------
__global__ void rm_mix_ln(const float* __restrict__ x, const float* __restrict__ g,
                          const float* __restrict__ beta, __bf16* __restrict__ xln) {
    int row = blockIdx.x;
    int b = row >> 3, t = row & 7;
    const float* xb = x + (size_t)b * TT * DD;
    int tid = threadIdx.x;
    float z[2];
#pragma unroll
    for (int e = 0; e < 2; ++e) {
        int d = tid + e * 256;
        int t2 = d >> 6, j = d & 63;
        z[e] = xb[(size_t)t * DD + d] + xb[(size_t)t2 * DD + t * 64 + j];
    }
    float s = z[0] + z[1];
    float q = z[0] * z[0] + z[1] * z[1];
#pragma unroll
    for (int off = 16; off > 0; off >>= 1) {
        s += __shfl_down(s, off);
        q += __shfl_down(q, off);
    }
    __shared__ float ps[8], pq[8];
    int w = tid >> 5, lane = tid & 31;
    if (lane == 0) { ps[w] = s; pq[w] = q; }
    __syncthreads();
    float S = 0.f, Q = 0.f;
#pragma unroll
    for (int i = 0; i < 8; ++i) { S += ps[i]; Q += pq[i]; }
    float mu = S * (1.0f / DD);
    float var = Q * (1.0f / DD) - mu * mu;
    float rstd = rsqrtf(var + 1e-5f);
#pragma unroll
    for (int e = 0; e < 2; ++e) {
        int d = tid + e * 256;
        float o = (z[e] - mu) * rstd * g[d] + beta[d];
        xln[(size_t)row * DD + d] = to_bf16(o);
    }
}

// ------------------------------------------------------------------
// Kernel C: fused per-token FFN: x_ln[64xD](TDM->LDS) -> h[64xH](LDS) -> y -> LN.
// Grid: (B/64, T). 512 threads = 16 waves, wave(wm in 0..3, wn in 0..3).
// ------------------------------------------------------------------
__global__ void __launch_bounds__(NTHR)
rm_ffn(const __bf16* __restrict__ xln, const __bf16* __restrict__ w1s,
       const __bf16* __restrict__ w2s, const float* __restrict__ b1,
       const float* __restrict__ b2, const float* __restrict__ fg,
       const float* __restrict__ fb, float* __restrict__ out) {
    extern __shared__ char smem[];
    __bf16* sA = (__bf16*)smem;                      // [BM][SA_STR]
    __bf16* sH = (__bf16*)(smem + SA_BYTES);         // [BM][SH_STR]
    float*  rs  = (float*)(smem + RED_OFF);          // [BM] row sums
    float*  rss = rs + BM;                           // [BM] row sumsq

    const int tid = threadIdx.x;
    const int t = blockIdx.y;
    const int b0 = blockIdx.x * BM;
    const int w = tid >> 5, lane = tid & 31;
    const int wm = w & 3, wn = w >> 2;
    const int l16 = lane & 15, hi = lane >> 4;
    const int m0 = wm * 16;

    // ---- Stage x_ln rows into LDS via the Tensor Data Mover ----
    // 2-D D#: tile 512 x 64 (elements x rows), elem = 2B, row stride = T*D elems,
    // pad_enable inserts 8 DWORDs (16 bf16) after every 256 DWORDs (one row)
    // -> LDS row stride becomes SA_STR = 528 bf16, matching manual layout.
    const unsigned long long ga =
        (unsigned long long)(const void*)(xln + ((size_t)b0 * TT + t) * DD);
#if defined(__has_builtin) && __has_builtin(__builtin_amdgcn_tensor_load_to_lds)
    if (w == 0) {
        u32x4 g0;
        g0[0] = 1u;                                           // count=1, user mode
        g0[1] = 0u;                                           // lds_addr = sA base
        g0[2] = (unsigned)(ga & 0xFFFFFFFFu);                 // global_addr[31:0]
        g0[3] = (unsigned)((ga >> 32) & 0x01FFFFFFu) | (2u << 30);  // addr hi | type=2
        i32x8 g1;
        g1[0] = (int)((1u << 16) | (1u << 20) | (7u << 22) | (7u << 25));
                 // data_size=2B, pad_enable, pad_interval=256 DW, pad_amount=8 DW
        g1[1] = (int)(512u << 16);   // tensor_dim0 = 512 (bits [79:48], low half)
        g1[2] = (int)(64u << 16);    // tensor_dim0 hi=0 | tensor_dim1 = 64
        g1[3] = (int)(512u << 16);   // tensor_dim1 hi=0 | tile_dim0 = 512
        g1[4] = 64;                  // tile_dim1 = 64, tile_dim2 = 0 (2-D)
        g1[5] = (int)(TT * DD);      // tensor_dim0_stride = 4096 elems
        g1[6] = 0;
        g1[7] = 0;
        i32x4 gz = {0, 0, 0, 0};
#if __clang_major__ >= 23
        i32x8 gz8 = {0, 0, 0, 0, 0, 0, 0, 0};
        __builtin_amdgcn_tensor_load_to_lds(g0, g1, gz, gz, gz8, 0);
#else
        __builtin_amdgcn_tensor_load_to_lds(g0, g1, gz, gz, 0);
#endif
        __builtin_amdgcn_s_wait_tensorcnt(0);
    }
#else
    for (int i = tid; i < BM * (DD / 8); i += NTHR) {
        int r = i >> 6, c = i & 63;
        const uint4* src = (const uint4*)(xln + ((size_t)(b0 + r) * TT + t) * DD + c * 8);
        *(uint4*)(sA + (size_t)r * SA_STR + c * 8) = *src;
    }
#endif
    __syncthreads();

    // ---- Phase 1: h = gelu(x_ln * W1 + b1), into sH (bf16) ----
#pragma unroll
    for (int half = 0; half < 2; ++half) {
        v8f acc[8] = {};
        for (int ks = 0; ks < DD / 32; ++ks) {
            v16bf a = load_a_frag(sA, SA_STR, m0 + l16, ks * 32, hi);
#pragma unroll
            for (int j = 0; j < 8; ++j) {
                int nt = wn * 16 + half * 8 + j;
                size_t ofs = ((((size_t)t * 16 + ks) * 64 + nt) * 32 + lane) * 16;
                v16bf bfr = load_b_frag(w1s, ofs);
                acc[j] = __builtin_amdgcn_wmma_f32_16x16x32_bf16(
                    false, a, false, bfr, (short)0, acc[j], false, false);
            }
        }
#pragma unroll
        for (int j = 0; j < 8; ++j) {
            int n = (wn * 16 + half * 8 + j) * 16 + l16;
            float b1v = b1[t * HH + n];
#pragma unroll
            for (int i = 0; i < 8; ++i) {
                float v = acc[j][i] + b1v;
                sH[(size_t)(m0 + hi * 8 + i) * SH_STR + n] = to_bf16(gelu_exact(v));
            }
        }
    }
    if (tid < BM) { rs[tid] = 0.f; rss[tid] = 0.f; }
    __syncthreads();

    // ---- Phase 2: y = h * W2 + b2; s = x_ln + y; LayerNorm(s) ----
    v8f acc2[8] = {};
    for (int ks = 0; ks < HH / 32; ++ks) {
        v16bf a = load_a_frag(sH, SH_STR, m0 + l16, ks * 32, hi);
#pragma unroll
        for (int j = 0; j < 8; ++j) {
            int nt = wn * 8 + j;
            size_t ofs = ((((size_t)t * 32 + ks) * 32 + nt) * 32 + lane) * 16;
            v16bf bfr = load_b_frag(w2s, ofs);
            acc2[j] = __builtin_amdgcn_wmma_f32_16x16x32_bf16(
                false, a, false, bfr, (short)0, acc2[j], false, false);
        }
    }
    float ls[8] = {}, lq[8] = {};
#pragma unroll
    for (int j = 0; j < 8; ++j) {
        int n = (wn * 8 + j) * 16 + l16;
        float b2v = b2[t * DD + n];
#pragma unroll
        for (int i = 0; i < 8; ++i) {
            int m = m0 + hi * 8 + i;
            float s = acc2[j][i] + b2v + from_bf16(sA[(size_t)m * SA_STR + n]);
            acc2[j][i] = s;
            ls[i] += s;
            lq[i] += s * s;
        }
    }
#pragma unroll
    for (int i = 0; i < 8; ++i) {
        int m = m0 + hi * 8 + i;
        atomicAdd(&rs[m], ls[i]);    // ds_add_f32
        atomicAdd(&rss[m], lq[i]);
    }
    __syncthreads();
#pragma unroll
    for (int i = 0; i < 8; ++i) {
        int m = m0 + hi * 8 + i;
        float mu = rs[m] * (1.0f / DD);
        float var = rss[m] * (1.0f / DD) - mu * mu;
        float rstd = rsqrtf(var + 1e-5f);
#pragma unroll
        for (int j = 0; j < 8; ++j) {
            int n = (wn * 8 + j) * 16 + l16;
            float o = (acc2[j][i] - mu) * rstd * fg[n] + fb[n];
            out[((size_t)(b0 + m) * TT + t) * DD + n] = o;
        }
    }
}

// ------------------------------------------------------------------
extern "C" void kernel_launch(void* const* d_in, const int* in_sizes, int n_in,
                              void* d_out, int out_size, void* d_ws, size_t ws_size,
                              hipStream_t stream) {
    const float* x    = (const float*)d_in[0];
    const float* W1   = (const float*)d_in[1];
    const float* b1   = (const float*)d_in[2];
    const float* W2   = (const float*)d_in[3];
    const float* b2   = (const float*)d_in[4];
    const float* tn_g = (const float*)d_in[5];
    const float* tn_b = (const float*)d_in[6];
    const float* fn_g = (const float*)d_in[7];
    const float* fn_b = (const float*)d_in[8];
    float* out = (float*)d_out;

    char* ws = (char*)d_ws;
    __bf16* xln = (__bf16*)(ws + WS_XLN_OFF);
    __bf16* w1s = (__bf16*)(ws + WS_W1S_OFF);
    __bf16* w2s = (__bf16*)(ws + WS_W2S_OFF);

    (void)in_sizes; (void)n_in; (void)out_size; (void)ws_size;

    hipFuncSetAttribute(reinterpret_cast<const void*>(rm_ffn),
                        hipFuncAttributeMaxDynamicSharedMemorySize, (int)SMEM_BYTES);

    rm_swizzle_w<<<(2 * TT * DD * HH) / 256, 256, 0, stream>>>(W1, W2, w1s, w2s);
    rm_mix_ln<<<ROWS, 256, 0, stream>>>(x, tn_g, tn_b, xln);
    rm_ffn<<<dim3(BB / BM, TT), NTHR, SMEM_BYTES, stream>>>(xln, w1s, w2s, b1, b2,
                                                            fn_g, fn_b, out);
}
